// GraphConvLayer_90692529422948
// MI455X (gfx1250) — matrix operand order repaired and checked by reading, not compiled, not run on hardware
//
#include <hip/hip_runtime.h>
#include <hip/hip_bf16.h>

// GNN layer for MI455X (gfx1250, wave32, WMMA).
//
// Refactor: concat-GEMMs split into per-half GEMMs so the per-edge 128x64
// matmul becomes a gather+add (saves ~12x FLOPs); all GEMMs run on
// v_wmma_f32_16x16x32_f16 with f32 accumulation. Node-level arrays
// (12.8 MB each) are L2-resident (192 MB L2); the edge stage is the
// bandwidth bottleneck (~620 MB of L2 gather+atomic traffic).

typedef __attribute__((ext_vector_type(16))) _Float16 v16h;
typedef __attribute__((ext_vector_type(8)))  _Float16 v8h;
typedef __attribute__((ext_vector_type(8)))  float    v8f;

#define DIM 64
// Transposed weight tile in LDS: WT[n * WT_STRIDE + k], k in [0,128).
// 136 halfwords = 272 B row stride: 16B-aligned rows + bank-conflict padding.
#define WT_STRIDE 136

#define WMMA_F16(A, B, C) \
  __builtin_amdgcn_wmma_f32_16x16x32_f16(false, (A), false, (B), (short)0, (C), false, false)

// ---- A fragment (16x32 f16, ISA 7.12.2 dense-A layout) -------------------
// lane l (half = l>>4, m = l&15): elements 0..7  = A[m][kBase + half*8 + i]
//                                 elements 8..15 = A[m][kBase + 16 + half*8 + i]
__device__ __forceinline__ v16h load_a_frag(const float* __restrict__ M,
                                            int rowBase, int kBase, int lane) {
  int half = lane >> 4, m = lane & 15;
  const float4* p = reinterpret_cast<const float4*>(
      M + (size_t)(rowBase + m) * DIM + kBase + half * 8);
  float4 x0 = p[0], x1 = p[1];   // k .. k+7
  float4 y0 = p[4], y1 = p[5];   // k+16 .. k+23
  v16h a;
  a[0]  = (_Float16)x0.x; a[1]  = (_Float16)x0.y; a[2]  = (_Float16)x0.z; a[3]  = (_Float16)x0.w;
  a[4]  = (_Float16)x1.x; a[5]  = (_Float16)x1.y; a[6]  = (_Float16)x1.z; a[7]  = (_Float16)x1.w;
  a[8]  = (_Float16)y0.x; a[9]  = (_Float16)y0.y; a[10] = (_Float16)y0.z; a[11] = (_Float16)y0.w;
  a[12] = (_Float16)y1.x; a[13] = (_Float16)y1.y; a[14] = (_Float16)y1.z; a[15] = (_Float16)y1.w;
  return a;
}

// ---- B fragment (32x16 f16, per ISA 16-bit B layout) ---------------------
// lane l: column n = l&15; elements 0..7 = B[kBase + (l>>4)*8 + i][n],
// elements 8..15 = B[kBase + 16 + (l>>4)*8 + i][n]
// WT is W transposed ([n][k]) so this is two 16-byte chunks -> 2x ds_load_b128.
__device__ __forceinline__ v16h load_b_frag(const _Float16* __restrict__ WT,
                                            int nt, int kBase, int lane) {
  int n = (nt << 4) + (lane & 15);
  const v8h* q = reinterpret_cast<const v8h*>(
      WT + n * WT_STRIDE + kBase + (lane >> 4) * 8);
  v8h lo = q[0], hi = q[2];
  return __builtin_shufflevector(lo, hi, 0,1,2,3,4,5,6,7,8,9,10,11,12,13,14,15);
}

// ---- Kernel 1: P_top = X@Wmsg[0:64], P_bot = X@Wmsg[64:128]; zero agg ----
__global__ void __launch_bounds__(256)
gnn_precompute(const float* __restrict__ X, const float* __restrict__ Wmsg,
               float* __restrict__ Ptop, float* __restrict__ Pbot,
               float* __restrict__ agg, int tiles) {
  __shared__ __align__(16) _Float16 WT[DIM * WT_STRIDE];
  for (int idx = threadIdx.x; idx < 2 * DIM * DIM; idx += blockDim.x) {
    int k = idx >> 6, n = idx & 63;                 // Wmsg row-major [128][64]
    WT[n * WT_STRIDE + k] = (_Float16)Wmsg[idx];
  }
  __syncthreads();

  int wave = threadIdx.x >> 5, lane = threadIdx.x & 31;
  int tile = blockIdx.x * 8 + wave;
  if (tile >= tiles) tile = tiles - 1;              // overshoot waves redo last tile
  int rowBase = tile << 4;

  // zero the aggregation buffer for this node tile (d_ws is poisoned 0xAA)
  float* aggTile = agg + (size_t)rowBase * DIM;
  for (int j = lane; j < 16 * DIM; j += 32) aggTile[j] = 0.0f;

  v16h a0 = load_a_frag(X, rowBase, 0, lane);
  v16h a1 = load_a_frag(X, rowBase, 32, lane);

  int m0 = (lane >> 4) * 8, n0 = lane & 15;
#pragma unroll
  for (int o = 0; o < 2; ++o) {                     // 0: W_top, 1: W_bot
    float* __restrict__ P = o ? Pbot : Ptop;
#pragma unroll
    for (int nt = 0; nt < 4; ++nt) {
      v16h b0 = load_b_frag(WT, nt, o * 64 + 0, lane);
      v16h b1 = load_b_frag(WT, nt, o * 64 + 32, lane);
      v8f c = {};
      c = WMMA_F16(a0, b0, c);
      c = WMMA_F16(a1, b1, c);
#pragma unroll
      for (int r = 0; r < 8; ++r)
        P[(size_t)(rowBase + m0 + r) * DIM + nt * 16 + n0] = c[r];
    }
  }
}

// ---- Kernel 2: per edge: agg[col] += ReLU(Ptop[row] + Pbot[col] + b) -----
__global__ void __launch_bounds__(256)
gnn_edge(const int* __restrict__ ei, const float* __restrict__ Ptop,
         const float* __restrict__ Pbot, const float* __restrict__ bmsg,
         float* __restrict__ agg, int E) {
  int t = blockIdx.x * 256 + threadIdx.x;           // E*16 threads, 4 floats each
  int e = t >> 4;
  if (e >= E) return;
  int c = (t & 15) * 4;
  int src = ei[e];                                  // edge_index[0][e]
  int dst = ei[E + e];                              // edge_index[1][e]
  float4 a  = *reinterpret_cast<const float4*>(Ptop + (size_t)src * DIM + c);
  float4 b  = *reinterpret_cast<const float4*>(Pbot + (size_t)dst * DIM + c);
  float4 bb = *reinterpret_cast<const float4*>(bmsg + c);
  float m0 = fmaxf(a.x + b.x + bb.x, 0.0f);
  float m1 = fmaxf(a.y + b.y + bb.y, 0.0f);
  float m2 = fmaxf(a.z + b.z + bb.z, 0.0f);
  float m3 = fmaxf(a.w + b.w + bb.w, 0.0f);
  float* dp = agg + (size_t)dst * DIM + c;
  atomicAdd(dp + 0, m0);
  atomicAdd(dp + 1, m1);
  atomicAdd(dp + 2, m2);
  atomicAdd(dp + 3, m3);
}

// ---- Kernel 3: out = LayerNorm(ReLU(X@Wu1 + agg@Wu2 + b)) * gamma + beta -
__global__ void __launch_bounds__(256)
gnn_update(const float* __restrict__ X, const float* __restrict__ agg,
           const float* __restrict__ Wupd, const float* __restrict__ bupd,
           const float* __restrict__ gamma, const float* __restrict__ beta,
           float* __restrict__ out, int tiles) {
  __shared__ __align__(16) _Float16 WT[DIM * WT_STRIDE];
  __shared__ __align__(16) float tilebuf[8][16 * DIM];  // 4 KB per wave
  for (int idx = threadIdx.x; idx < 2 * DIM * DIM; idx += blockDim.x) {
    int k = idx >> 6, n = idx & 63;
    WT[n * WT_STRIDE + k] = (_Float16)Wupd[idx];
  }
  __syncthreads();

  int wave = threadIdx.x >> 5, lane = threadIdx.x & 31;
  int tile = blockIdx.x * 8 + wave;
  if (tile >= tiles) tile = tiles - 1;
  int rowBase = tile << 4;

  v16h ax0 = load_a_frag(X,   rowBase, 0,  lane);
  v16h ax1 = load_a_frag(X,   rowBase, 32, lane);
  v16h ag0 = load_a_frag(agg, rowBase, 0,  lane);
  v16h ag1 = load_a_frag(agg, rowBase, 32, lane);

  int m0 = (lane >> 4) * 8, n0 = lane & 15;
  float* buf = tilebuf[wave];
#pragma unroll
  for (int nt = 0; nt < 4; ++nt) {
    v16h b0 = load_b_frag(WT, nt, 0,  lane);   // W_u1 K=0..31
    v16h b1 = load_b_frag(WT, nt, 32, lane);   // W_u1 K=32..63
    v16h b2 = load_b_frag(WT, nt, 64, lane);   // W_u2 K=0..31
    v16h b3 = load_b_frag(WT, nt, 96, lane);   // W_u2 K=32..63
    v8f c = {};
    c = WMMA_F16(ax0, b0, c);
    c = WMMA_F16(ax1, b1, c);
    c = WMMA_F16(ag0, b2, c);
    c = WMMA_F16(ag1, b3, c);
    float bias = bupd[nt * 16 + n0];
#pragma unroll
    for (int r = 0; r < 8; ++r)
      buf[(m0 + r) * DIM + nt * 16 + n0] = fmaxf(c[r] + bias, 0.0f);
  }
  __syncthreads();

  // LayerNorm: lanes 0..15 each own one row of the 16x64 tile.
  if (lane < 16) {
    const float4* rowp = reinterpret_cast<const float4*>(buf + lane * DIM);
    float s = 0.0f, ss = 0.0f;
#pragma unroll
    for (int j = 0; j < 16; ++j) {
      float4 v = rowp[j];
      s  += v.x + v.y + v.z + v.w;
      ss += v.x * v.x + v.y * v.y + v.z * v.z + v.w * v.w;
    }
    float mean = s * (1.0f / 64.0f);
    float var  = ss * (1.0f / 64.0f) - mean * mean;
    float rstd = rsqrtf(var + 1e-5f);
    float4* op = reinterpret_cast<float4*>(out + (size_t)(rowBase + lane) * DIM);
#pragma unroll
    for (int j = 0; j < 16; ++j) {
      float4 v  = rowp[j];
      float4 g  = reinterpret_cast<const float4*>(gamma)[j];
      float4 be = reinterpret_cast<const float4*>(beta)[j];
      float4 r;
      r.x = (v.x - mean) * rstd * g.x + be.x;
      r.y = (v.y - mean) * rstd * g.y + be.y;
      r.z = (v.z - mean) * rstd * g.z + be.z;
      r.w = (v.w - mean) * rstd * g.w + be.w;
      op[j] = r;
    }
  }
}

extern "C" void kernel_launch(void* const* d_in, const int* in_sizes, int n_in,
                              void* d_out, int out_size, void* d_ws, size_t ws_size,
                              hipStream_t stream) {
  const float* X     = (const float*)d_in[0];   // [N,64]
  const int*   ei    = (const int*)  d_in[1];   // [2,E]
  const float* Wmsg  = (const float*)d_in[2];   // [128,64]
  const float* bmsg  = (const float*)d_in[3];   // [64]
  const float* Wupd  = (const float*)d_in[4];   // [128,64]
  const float* bupd  = (const float*)d_in[5];   // [64]
  const float* gamma = (const float*)d_in[6];   // [64]
  const float* beta  = (const float*)d_in[7];   // [64]
  float* out = (float*)d_out;

  int N = in_sizes[0] / DIM;
  int E = in_sizes[1] / 2;

  // Workspace: Ptop | Pbot | agg, each N*64 f32 (3 * 12.8 MB).
  float* Ptop = (float*)d_ws;
  float* Pbot = Ptop + (size_t)N * DIM;
  float* agg  = Pbot + (size_t)N * DIM;

  int tiles   = (N + 15) / 16;        // N = 50000 -> 3125 tiles (N % 16 == 0)
  int blocksT = (tiles + 7) / 8;      // 8 waves (tiles) per 256-thread block

  gnn_precompute<<<blocksT, 256, 0, stream>>>(X, Wmsg, Ptop, Pbot, agg, tiles);

  long long eThreads = (long long)E * 16;
  int eBlocks = (int)((eThreads + 255) / 256);
  gnn_edge<<<eBlocks, 256, 0, stream>>>(ei, Ptop, Pbot, bmsg, agg, E);

  gnn_update<<<blocksT, 256, 0, stream>>>(X, agg, Wupd, bupd, gamma, beta, out, tiles);
}
